// MultiHeadSelfAttention_6897717477661
// MI455X (gfx1250) — matrix operand order repaired and checked by reading, not compiled, not run on hardware
//
#include <hip/hip_runtime.h>
#include <hip/hip_bf16.h>
#include <stdint.h>

// ---------------------------------------------------------------------------
// MI455X (gfx1250) multi-head self-attention, f16 WMMA everywhere.
// B=4, S=2048, D=1024, H=16, DH=64.
// ---------------------------------------------------------------------------

typedef _Float16 half_t;
typedef __attribute__((ext_vector_type(16))) _Float16 v16h;
typedef __attribute__((ext_vector_type(8)))  float    v8f;

union Frag16 { v16h v; uint32_t u[8]; };

// A-matrix (16x32 f16) K-pair base for VGPR i, lane-half hi (contiguous k = 16*hi + 2*i, matching B).
__device__ __forceinline__ int a_kpair(int i, int hi) {
    return 2 * i + 16 * hi;
}

#define SEQ   2048
#define DMODEL 1024
#define NHEAD 16
#define DHEAD 64
#define BATCH 4
#define MROWS (BATCH * SEQ)      // 8192

// ---------------------------------------------------------------------------
// fp32 -> fp16 conversion
// ---------------------------------------------------------------------------
__global__ __launch_bounds__(256) void cvt_f32_f16_kernel(
    const float* __restrict__ src, half_t* __restrict__ dst, int n) {
    int i = blockIdx.x * 256 + threadIdx.x;
    if (i < n) dst[i] = (half_t)src[i];
}

// ---------------------------------------------------------------------------
// QKV projection: out[e] = sum_d x[m,d] * W[e,d]  (x @ W^T)
// grid = (M/128, N/64, 3), block = 256 (8 waves). Wave owns 16(M) x 64(N).
// Output layout: qkv[which][b*H+h][s][dh]  (f16)
// ---------------------------------------------------------------------------
__global__ __launch_bounds__(256) void qkv_gemm_kernel(
    const half_t* __restrict__ xh,   // [8192, 1024]
    const half_t* __restrict__ w3,   // [3, 1024, 1024]
    half_t* __restrict__ qkv)        // [3, 64, 2048, 64]
{
    const int wave = threadIdx.x >> 5;
    const int lane = threadIdx.x & 31;
    const int nl   = lane & 15;
    const int hi   = lane >> 4;
    const int m0   = blockIdx.x * 128 + wave * 16;
    const int n0   = blockIdx.y * 64;
    const int which = blockIdx.z;
    const half_t* W = w3 + (size_t)which * DMODEL * DMODEL;

    v8f acc[4] = {};
    const int mrow = m0 + nl;
    const half_t* xrow = xh + (size_t)mrow * DMODEL;

    for (int k0 = 0; k0 < DMODEL; k0 += 32) {
        Frag16 a;
#pragma unroll
        for (int i = 0; i < 8; ++i)
            a.u[i] = *reinterpret_cast<const uint32_t*>(xrow + k0 + a_kpair(i, hi));
#pragma unroll
        for (int nt = 0; nt < 4; ++nt) {
            Frag16 b;
            const half_t* wrow = W + (size_t)(n0 + nt * 16 + nl) * DMODEL + k0;
#pragma unroll
            for (int i = 0; i < 8; ++i)
                b.u[i] = *reinterpret_cast<const uint32_t*>(wrow + 2 * i + 16 * hi);
            acc[nt] = __builtin_amdgcn_wmma_f32_16x16x32_f16(
                false, a.v, false, b.v, (short)0, acc[nt], false, false);
        }
    }

#pragma unroll
    for (int nt = 0; nt < 4; ++nt) {
#pragma unroll
        for (int i = 0; i < 8; ++i) {
            int m = m0 + i + 8 * hi;
            int n = n0 + nt * 16 + nl;
            int b = m >> 11, s = m & (SEQ - 1);
            int h = n >> 6, dh = n & 63;
            size_t dst = (((size_t)which * (BATCH * NHEAD) + b * NHEAD + h) * SEQ + s) * DHEAD + dh;
            qkv[dst] = (half_t)acc[nt][i];
        }
    }
}

// ---------------------------------------------------------------------------
// RoPE in place on q,k (layout [B*H, S, 64]); interleaved pairs.
// grid = (B*H*S*32/256, 2)
// ---------------------------------------------------------------------------
__global__ __launch_bounds__(256) void rope_kernel(
    half_t* __restrict__ q, half_t* __restrict__ k, const int* __restrict__ tokpos)
{
    int tid = blockIdx.x * 256 + threadIdx.x;      // pair index
    half_t* p = (blockIdx.y == 0) ? q : k;
    int f   = tid & 31;
    int row = tid >> 5;                             // (b*H + h)*S + s
    int s   = row & (SEQ - 1);
    int b   = row >> 15;                            // H*S = 2^15
    int pos = tokpos[b * SEQ + s];

    float freq = __expf(-9.210340371976184f * ((float)f * (1.0f / 32.0f)));
    float ang = (float)pos * freq;
    float sn, cs;
    __sincosf(ang, &sn, &cs);

    size_t base = (size_t)row * DHEAD + 2 * f;
    float x1 = (float)p[base];
    float x2 = (float)p[base + 1];
    p[base]     = (half_t)(x1 * cs - x2 * sn);
    p[base + 1] = (half_t)(x1 * sn + x2 * cs);
}

// ---------------------------------------------------------------------------
// Flash attention (causal). grid = (S/128, B*H), block = 256 (8 waves).
// Wave owns 16 q-rows; K/V streamed through LDS in 64-key tiles.
// K tile uses gfx1250 async global->LDS DMA (ASYNCcnt); V transposes through
// VGPRs; next tile prefetched with global_prefetch. Output [B, S, H*64] f16.
// ---------------------------------------------------------------------------
__global__ __launch_bounds__(256) void flash_attn_kernel(
    const half_t* __restrict__ q, const half_t* __restrict__ k,
    const half_t* __restrict__ v, half_t* __restrict__ o)
{
    __shared__ half_t Ks[64 * 72];        // keys   [t][dh], padded
    __shared__ half_t Vt[64 * 72];        // values [dh][t], transposed, padded
    __shared__ half_t Ps[8 * 16 * 72];    // per-wave P staging [m][t], padded

    const int wave = threadIdx.x >> 5;
    const int lane = threadIdx.x & 31;
    const int nl   = lane & 15;
    const int hi   = lane >> 4;
    const int bh   = blockIdx.y;
    const int b    = bh >> 4;
    const int h    = bh & 15;
    const int q0   = blockIdx.x * 128;
    const int qw   = q0 + wave * 16;

    const half_t* qbase = q + (size_t)bh * SEQ * DHEAD;
    const half_t* kbase = k + (size_t)bh * SEQ * DHEAD;
    const half_t* vbase = v + (size_t)bh * SEQ * DHEAD;

    // Q fragments (A layout), resident for the whole kernel.
    Frag16 aq0, aq1;
    {
        const half_t* qr = qbase + (size_t)(qw + nl) * DHEAD;
#pragma unroll
        for (int i = 0; i < 8; ++i) {
            int kp = a_kpair(i, hi);
            aq0.u[i] = *reinterpret_cast<const uint32_t*>(qr + kp);
            aq1.u[i] = *reinterpret_cast<const uint32_t*>(qr + 32 + kp);
        }
    }

    v8f acc[4] = {};
    float mprev[8], lsum[8];
#pragma unroll
    for (int i = 0; i < 8; ++i) { mprev[i] = -INFINITY; lsum[i] = 0.0f; }

    half_t* Pw = Ps + wave * 16 * 72;
    const int jmax = (q0 + 127) >> 6;

    for (int j = 0; j <= jmax; ++j) {
        const int jb = j * 64;
        __syncthreads();
        {   // cooperative K/V tile load
            int r = threadIdx.x >> 2;
            int c = (threadIdx.x & 3) * 16;
            const half_t* krow = kbase + (size_t)(jb + r) * DHEAD + c;
            const half_t* vrow = vbase + (size_t)(jb + r) * DHEAD + c;

            // K tile: async DMA global -> LDS, 2 x 16B per lane; INST_OFFSET
            // applies to both the global and LDS addresses (ISA 08 §4.4).
            uint32_t lds_k = (uint32_t)(uintptr_t)&Ks[r * 72 + c];
            uint64_t g_k   = (uint64_t)(uintptr_t)krow;
            asm volatile("global_load_async_to_lds_b128 %0, %1, off"
                         :: "v"(lds_k), "v"(g_k) : "memory");
            asm volatile("global_load_async_to_lds_b128 %0, %1, off offset:16"
                         :: "v"(lds_k), "v"(g_k) : "memory");

            // V tile: VGPR round-trip, stored transposed for PV B-fragments.
#pragma unroll
            for (int x = 0; x < 16; ++x)
                Vt[(c + x) * 72 + r] = vrow[x];

            // Prefetch next tile while this one is consumed.
            if (j < jmax) {
                __builtin_prefetch(krow + 64 * DHEAD, 0, 1);
                __builtin_prefetch(vrow + 64 * DHEAD, 0, 1);
            }
            asm volatile("s_wait_asynccnt 0x0" ::: "memory");
        }
        __syncthreads();

        // scores: S = Q * K^T  (B fragment: N = key index, K = dh)
        v8f sc[4];
#pragma unroll
        for (int nt = 0; nt < 4; ++nt) {
            Frag16 b0, b1;
            const int t = nt * 16 + nl;
#pragma unroll
            for (int i = 0; i < 8; ++i) {
                int dh = 2 * i + 16 * hi;
                b0.u[i] = *reinterpret_cast<const uint32_t*>(&Ks[t * 72 + dh]);
                b1.u[i] = *reinterpret_cast<const uint32_t*>(&Ks[t * 72 + dh + 32]);
            }
            v8f z = {};
            z = __builtin_amdgcn_wmma_f32_16x16x32_f16(false, aq0.v, false, b0.v, (short)0, z, false, false);
            sc[nt] = __builtin_amdgcn_wmma_f32_16x16x32_f16(false, aq1.v, false, b1.v, (short)0, z, false, false);
        }

        // online softmax update (row reductions across 16 lanes, wave32)
#pragma unroll
        for (int i = 0; i < 8; ++i) {
            const int qrow = qw + i + 8 * hi;
            float mx = -INFINITY;
#pragma unroll
            for (int nt = 0; nt < 4; ++nt) {
                int col = jb + nt * 16 + nl;
                float sv = sc[nt][i] * 0.125f;
                sv = (col <= qrow) ? sv : -INFINITY;
                sc[nt][i] = sv;
                mx = fmaxf(mx, sv);
            }
            mx = fmaxf(mx, __shfl_xor(mx, 1));
            mx = fmaxf(mx, __shfl_xor(mx, 2));
            mx = fmaxf(mx, __shfl_xor(mx, 4));
            mx = fmaxf(mx, __shfl_xor(mx, 8));
            float mnew = fmaxf(mprev[i], mx);
            float rs = 0.0f;
#pragma unroll
            for (int nt = 0; nt < 4; ++nt) {
                float pexp = __expf(sc[nt][i] - mnew);
                sc[nt][i] = pexp;
                rs += pexp;
            }
            rs += __shfl_xor(rs, 1);
            rs += __shfl_xor(rs, 2);
            rs += __shfl_xor(rs, 4);
            rs += __shfl_xor(rs, 8);
            float scalef = __expf(mprev[i] - mnew);
            lsum[i] = lsum[i] * scalef + rs;
            mprev[i] = mnew;
#pragma unroll
            for (int nt = 0; nt < 4; ++nt) acc[nt][i] *= scalef;
        }

        // stage P (C-frag layout -> LDS row-major) for A-fragment reload
#pragma unroll
        for (int nt = 0; nt < 4; ++nt)
#pragma unroll
            for (int i = 0; i < 8; ++i)
                Pw[(i + 8 * hi) * 72 + nt * 16 + nl] = (half_t)sc[nt][i];
        asm volatile("s_wait_dscnt 0" ::: "memory");   // same-wave LDS RAW

        // O += P * V  (A from Pw, B from transposed Vt)
        Frag16 ap0, ap1;
#pragma unroll
        for (int i = 0; i < 8; ++i) {
            int kp = a_kpair(i, hi);
            ap0.u[i] = *reinterpret_cast<const uint32_t*>(&Pw[nl * 72 + kp]);
            ap1.u[i] = *reinterpret_cast<const uint32_t*>(&Pw[nl * 72 + 32 + kp]);
        }
#pragma unroll
        for (int nt = 0; nt < 4; ++nt) {
            Frag16 b0, b1;
            const int dh = nt * 16 + nl;
#pragma unroll
            for (int i = 0; i < 8; ++i) {
                int t = 2 * i + 16 * hi;
                b0.u[i] = *reinterpret_cast<const uint32_t*>(&Vt[dh * 72 + t]);
                b1.u[i] = *reinterpret_cast<const uint32_t*>(&Vt[dh * 72 + t + 32]);
            }
            acc[nt] = __builtin_amdgcn_wmma_f32_16x16x32_f16(false, ap0.v, false, b0.v, (short)0, acc[nt], false, false);
            acc[nt] = __builtin_amdgcn_wmma_f32_16x16x32_f16(false, ap1.v, false, b1.v, (short)0, acc[nt], false, false);
        }
    }

    // normalize + store to [B, S, H*DH]
#pragma unroll
    for (int i = 0; i < 8; ++i) {
        float inv = 1.0f / lsum[i];
#pragma unroll
        for (int nt = 0; nt < 4; ++nt) acc[nt][i] *= inv;
    }
#pragma unroll
    for (int nt = 0; nt < 4; ++nt)
#pragma unroll
        for (int i = 0; i < 8; ++i) {
            int srow = qw + i + 8 * hi;
            int dh = nt * 16 + nl;
            o[((size_t)b * SEQ + srow) * DMODEL + h * DHEAD + dh] = (half_t)acc[nt][i];
        }
}

// ---------------------------------------------------------------------------
// Output projection: final[m,n] = sum_d attn[m,d] * Wo[n,d], f32 out.
// ---------------------------------------------------------------------------
__global__ __launch_bounds__(256) void out_gemm_kernel(
    const half_t* __restrict__ ain,  // [8192, 1024]
    const half_t* __restrict__ W,    // [1024, 1024]
    float* __restrict__ out)         // [8192, 1024]
{
    const int wave = threadIdx.x >> 5;
    const int lane = threadIdx.x & 31;
    const int nl   = lane & 15;
    const int hi   = lane >> 4;
    const int m0   = blockIdx.x * 128 + wave * 16;
    const int n0   = blockIdx.y * 64;

    v8f acc[4] = {};
    const half_t* arow = ain + (size_t)(m0 + nl) * DMODEL;

    for (int k0 = 0; k0 < DMODEL; k0 += 32) {
        Frag16 a;
#pragma unroll
        for (int i = 0; i < 8; ++i)
            a.u[i] = *reinterpret_cast<const uint32_t*>(arow + k0 + a_kpair(i, hi));
#pragma unroll
        for (int nt = 0; nt < 4; ++nt) {
            Frag16 b;
            const half_t* wrow = W + (size_t)(n0 + nt * 16 + nl) * DMODEL + k0;
#pragma unroll
            for (int i = 0; i < 8; ++i)
                b.u[i] = *reinterpret_cast<const uint32_t*>(wrow + 2 * i + 16 * hi);
            acc[nt] = __builtin_amdgcn_wmma_f32_16x16x32_f16(
                false, a.v, false, b.v, (short)0, acc[nt], false, false);
        }
    }
#pragma unroll
    for (int nt = 0; nt < 4; ++nt)
#pragma unroll
        for (int i = 0; i < 8; ++i) {
            int m = m0 + i + 8 * hi;
            int n = n0 + nt * 16 + nl;
            out[(size_t)m * DMODEL + n] = acc[nt][i];
        }
}

// ---------------------------------------------------------------------------
extern "C" void kernel_launch(void* const* d_in, const int* in_sizes, int n_in,
                              void* d_out, int out_size, void* d_ws, size_t ws_size,
                              hipStream_t stream) {
    const float* x      = (const float*)d_in[0];
    const int*   tokpos = (const int*)d_in[1];
    const float* Wq     = (const float*)d_in[2];
    const float* Wk     = (const float*)d_in[3];
    const float* Wv     = (const float*)d_in[4];
    const float* Wo     = (const float*)d_in[5];
    float* out = (float*)d_out;

    const size_t MD = (size_t)MROWS * DMODEL;    // 8M elems
    const size_t WD = (size_t)DMODEL * DMODEL;   // 1M elems

    half_t* ws   = (half_t*)d_ws;
    half_t* xh   = ws;                  // 8M halves
    half_t* w3   = xh + MD;             // 3M (Wq,Wk,Wv stacked)
    half_t* woh  = w3 + 3 * WD;         // 1M
    half_t* qbuf = woh + WD;            // 8M (q) -- q,k,v contiguous = [3][...]
    half_t* kbuf = qbuf + MD;           // 8M (k)
    half_t* vbuf = kbuf + MD;           // 8M (v)
    half_t* attn = vbuf + MD;           // 8M

    cvt_f32_f16_kernel<<<(int)(MD / 256), 256, 0, stream>>>(x, xh, (int)MD);
    cvt_f32_f16_kernel<<<(int)(WD / 256), 256, 0, stream>>>(Wq, w3, (int)WD);
    cvt_f32_f16_kernel<<<(int)(WD / 256), 256, 0, stream>>>(Wk, w3 + WD, (int)WD);
    cvt_f32_f16_kernel<<<(int)(WD / 256), 256, 0, stream>>>(Wv, w3 + 2 * WD, (int)WD);
    cvt_f32_f16_kernel<<<(int)(WD / 256), 256, 0, stream>>>(Wo, woh, (int)WD);

    qkv_gemm_kernel<<<dim3(MROWS / 128, DMODEL / 64, 3), 256, 0, stream>>>(xh, w3, qbuf);
    rope_kernel<<<dim3((BATCH * NHEAD * SEQ * 32) / 256, 2), 256, 0, stream>>>(qbuf, kbuf, tokpos);
    flash_attn_kernel<<<dim3(SEQ / 128, BATCH * NHEAD), 256, 0, stream>>>(qbuf, kbuf, vbuf, attn);
    out_gemm_kernel<<<dim3(MROWS / 128, DMODEL / 64), 256, 0, stream>>>(attn, woh, out);
}
